// LongformerICD_63617055588567
// MI455X (gfx1250) — matrix operand order repaired and checked by reading, not compiled, not run on hardware
//
#include <hip/hip_runtime.h>
#include <hip/hip_bf16.h>
#include <math.h>

// ---------------- problem constants ----------------
#define Bz 2
#define Sz 2048
#define Dz 768
#define Hz 12
#define Lz 2
#define DHz 64
#define Wz 256
#define FFz 3072
#define NUM_LABELS 50
#define NCz (Sz / Wz)            // 8 chunks
#define BSD ((size_t)Bz * Sz * Dz)

// ---------------- WMMA / TDM types ----------------
typedef __attribute__((ext_vector_type(16))) _Float16 v16h;
typedef __attribute__((ext_vector_type(4)))  _Float16 v4h;
typedef __attribute__((ext_vector_type(8)))  float    v8f;
typedef __attribute__((ext_vector_type(4)))  unsigned int u32x4;
typedef __attribute__((ext_vector_type(4)))  unsigned int tdm_g0_t;
typedef __attribute__((ext_vector_type(8)))  int          tdm_g1_t;
typedef __attribute__((ext_vector_type(4)))  int          tdm_g23_t;

union Frag16 { v16h v; u32x4 q[2]; };

// Tensor Data Mover availability (device pass only; host pass falls back)
#if defined(__HIP_DEVICE_COMPILE__) && defined(__has_builtin)
#if __has_builtin(__builtin_amdgcn_tensor_load_to_lds) && __has_builtin(__builtin_amdgcn_s_wait_tensorcnt)
#define USE_TDM 1
#endif
#endif

// Generic shared-memory pointer -> 32-bit LDS byte offset.
// (On AMDGPU the generic form of an LDS address carries the LDS offset in
//  its low 32 bits; see CDNA5 ISA aperture mapping.)
static __device__ __forceinline__ unsigned lds_off_u32(const void* p) {
    return (unsigned)(unsigned long long)(size_t)p;
}

// =====================================================================
// Generic GEMM:  C[M,N] = A[M,K] @ Wt[K,N] + bias[N]
//   * A tile (128x32 f32) pulled into LDS by the Tensor Data Mover
//     (2-D D# descriptor, one issue per block per K-step, TENSORcnt wait)
//   * B tile staged manually: coalesced float4 loads, f32->f16 convert,
//     transposed [n][k] f16 layout so each fragment is 2x ds_load_b128
//   * compute: v_wmma_f32_16x16x32_f16, wave tile 32x32 (2x2 WMMA)
//   * f32->f16 of A happens at fragment load (VALU co-executes with WMMA)
// M%128==0, N%64==0, K%32==0 hold for all GEMMs here.
// =====================================================================
#define TM 128
#define TN 64
#define TK 32
#define BSTR (TK + 8)   // transposed-B row stride (halfs); 80B, 16B aligned

__global__ __launch_bounds__(256)
void gemm_wmma_kernel(const float* __restrict__ A, const float* __restrict__ Wt,
                      const float* __restrict__ bias, float* __restrict__ C,
                      int M, int N, int K, int applyGelu)
{
    __shared__ float    As32[TM * TK];     // 16 KB, f32, row stride 32 (128B)
    __shared__ _Float16 Bs[TN * BSTR];     // 5 KB, f16, transposed [n][k]

    const int tid  = threadIdx.x;
    const int lane = tid & 31;
    const int wave = tid >> 5;
    const int wm   = (wave & 3) * 32;
    const int wn   = (wave >> 2) * 32;
    const int gm0  = blockIdx.y * TM;
    const int gn0  = blockIdx.x * TN;

    v8f acc[2][2] = {};

    const int arow = lane & 15;
    const int kbA  = (lane >> 4) * 8;      // A K sub-block {0,8}
    const int kbB  = (lane >> 4) * 8;      // B K sub-block {0,8}

#ifdef USE_TDM
    const unsigned ldsA = lds_off_u32(As32);
    const unsigned long long abase =
        (unsigned long long)(A + (size_t)gm0 * K);   // byte address of row gm0
#endif

    for (int k0 = 0; k0 < K; k0 += TK) {
        // ---------------- A tile: Tensor Data Mover ----------------
#ifdef USE_TDM
        if (tid == 0) {
            const unsigned long long ga = abase + (unsigned long long)k0 * 4ull;
            // D# group 0: count=1 | lds_addr | global_addr | type=2
            tdm_g0_t g0 = { 1u, ldsA, (unsigned)ga,
                            (unsigned)(ga >> 32) | 0x80000000u };
            // D# group 1: data_size=4B, tensor (K x M), tile (32 x 128),
            // tensor_dim0_stride = K elements
            tdm_g1_t g1 = { (int)(2u << 16),
                            (int)((K & 0xFFFF) << 16),
                            (int)(((K >> 16) & 0xFFFF) | ((M & 0xFFFF) << 16)),
                            (int)(((M >> 16) & 0xFFFF) | (TK << 16)),
                            (int)TM,
                            (int)K,
                            0, 0 };
            tdm_g23_t gz = { 0, 0, 0, 0 };
#if __clang_major__ >= 23
            tdm_g1_t z8 = { 0, 0, 0, 0, 0, 0, 0, 0 };
            __builtin_amdgcn_tensor_load_to_lds(g0, g1, gz, gz, z8, 0);
#else
            __builtin_amdgcn_tensor_load_to_lds(g0, g1, gz, gz, 0);
#endif
        }
#else
        {   // fallback: vectorized manual stage (gather then store)
            float4 areg[4];
            #pragma unroll
            for (int it = 0; it < 4; ++it) {
                int idx = it * 256 + tid;            // 1024 float4 slots
                int r = idx >> 3, c4 = idx & 7;
                areg[it] = *(const float4*)(&A[(size_t)(gm0 + r) * K + k0 + c4 * 4]);
            }
            #pragma unroll
            for (int it = 0; it < 4; ++it) {
                int idx = it * 256 + tid;
                int r = idx >> 3, c4 = idx & 7;
                *(float4*)(&As32[r * TK + c4 * 4]) = areg[it];
            }
        }
#endif
        // ---------------- B tile: coalesced loads, f16 transpose ----------------
        float4 breg[2];
        #pragma unroll
        for (int it = 0; it < 2; ++it) {
            int idx = it * 256 + tid;                // 512 float4 slots (32x64)
            int kk = idx >> 4, n4 = idx & 15;
            breg[it] = *(const float4*)(&Wt[(size_t)(k0 + kk) * N + gn0 + n4 * 4]);
        }
        if (k0 + TK < K)  // next weight tile prefetch -> global_prefetch_b8
            __builtin_prefetch(&Wt[(size_t)(k0 + TK + (tid >> 4)) * N + gn0], 0, 1);
        #pragma unroll
        for (int it = 0; it < 2; ++it) {
            int idx = it * 256 + tid;
            int kk = idx >> 4, n4 = idx & 15;
            Bs[(n4 * 4 + 0) * BSTR + kk] = (_Float16)breg[it].x;
            Bs[(n4 * 4 + 1) * BSTR + kk] = (_Float16)breg[it].y;
            Bs[(n4 * 4 + 2) * BSTR + kk] = (_Float16)breg[it].z;
            Bs[(n4 * 4 + 3) * BSTR + kk] = (_Float16)breg[it].w;
        }
#ifdef USE_TDM
        __builtin_amdgcn_s_wait_tensorcnt(0);        // TDM done (issuing wave)
#endif
        __syncthreads();

        // ---------------- fragments ----------------
        Frag16 a[2], b[2];
        #pragma unroll
        for (int i = 0; i < 2; ++i) {
            const float* p = &As32[(wm + i * 16 + arow) * TK];
            float4 f0 = *(const float4*)(p + kbA);
            float4 f1 = *(const float4*)(p + kbA + 4);
            float4 f2 = *(const float4*)(p + 16 + kbA);
            float4 f3 = *(const float4*)(p + 16 + kbA + 4);
            v16h av;
            av[0]  = (_Float16)f0.x; av[1]  = (_Float16)f0.y;
            av[2]  = (_Float16)f0.z; av[3]  = (_Float16)f0.w;
            av[4]  = (_Float16)f1.x; av[5]  = (_Float16)f1.y;
            av[6]  = (_Float16)f1.z; av[7]  = (_Float16)f1.w;
            av[8]  = (_Float16)f2.x; av[9]  = (_Float16)f2.y;
            av[10] = (_Float16)f2.z; av[11] = (_Float16)f2.w;
            av[12] = (_Float16)f3.x; av[13] = (_Float16)f3.y;
            av[14] = (_Float16)f3.z; av[15] = (_Float16)f3.w;
            a[i].v = av;
        }
        #pragma unroll
        for (int j = 0; j < 2; ++j) {
            const _Float16* p = &Bs[(wn + j * 16 + arow) * BSTR + kbB];
            b[j].q[0] = *(const u32x4*)(p);
            b[j].q[1] = *(const u32x4*)(p + 16);
        }
        #pragma unroll
        for (int i = 0; i < 2; ++i)
            #pragma unroll
            for (int j = 0; j < 2; ++j)
                acc[i][j] = __builtin_amdgcn_wmma_f32_16x16x32_f16(
                    false, a[i].v, false, b[j].v, (short)0, acc[i][j], false, false);
        __syncthreads();
    }

    // ---- epilogue: C/D layout M = r + 8*(lane>>4), N = lane&15 ----
    const int mbase = 8 * (lane >> 4);
    const int ncol  = lane & 15;
    #pragma unroll
    for (int i = 0; i < 2; ++i) {
        #pragma unroll
        for (int j = 0; j < 2; ++j) {
            #pragma unroll
            for (int r = 0; r < 8; ++r) {
                int m = gm0 + wm + i * 16 + r + mbase;
                int n = gn0 + wn + j * 16 + ncol;
                float v = acc[i][j][r] + bias[n];
                if (applyGelu) v = 0.5f * v * (1.f + erff(v * 0.70710678118f));
                C[(size_t)m * N + n] = v;
            }
        }
    }
}

// =====================================================================
// Sliding-window attention: one (b,h,chunk) per block, one query row per
// thread (W==256==blockDim). Online softmax over {key0} U window keys.
// =====================================================================
__global__ __launch_bounds__(256)
void sliding_attn_kernel(const float* __restrict__ Q, const float* __restrict__ Kb,
                         const float* __restrict__ Vb, const int* __restrict__ am,
                         float* __restrict__ AO)
{
    const int blk = blockIdx.x;
    const int c = blk % NCz;
    const int h = (blk / NCz) % Hz;
    const int b = blk / (NCz * Hz);
    const int tid = threadIdx.x;
    const int qrow = c * Wz + tid;
    const float scale = 0.125f;  // 1/sqrt(64)

    const float4* q4 = (const float4*)(Q + ((size_t)(b * Sz + qrow)) * Dz + h * DHz);
    float4 qv[16];
    #pragma unroll
    for (int d = 0; d < 16; ++d) qv[d] = q4[d];

    float mM = -3.4e38f, sS = 0.f;
    float4 ov[16];
    #pragma unroll
    for (int d = 0; d < 16; ++d) ov[d] = make_float4(0.f, 0.f, 0.f, 0.f);

    auto accum = [&](int kp) {
        const float4* k4 = (const float4*)(Kb + ((size_t)(b * Sz + kp)) * Dz + h * DHz);
        const float4* v4 = (const float4*)(Vb + ((size_t)(b * Sz + kp)) * Dz + h * DHz);
        float dot = 0.f;
        #pragma unroll
        for (int d = 0; d < 16; ++d) {
            float4 kk = k4[d];
            dot += qv[d].x * kk.x + qv[d].y * kk.y + qv[d].z * kk.z + qv[d].w * kk.w;
        }
        float sc = dot * scale;
        float nm = fmaxf(mM, sc);
        float corr = __expf(mM - nm);
        float pe   = __expf(sc - nm);
        sS = sS * corr + pe;
        #pragma unroll
        for (int d = 0; d < 16; ++d) {
            float4 vv = v4[d];
            ov[d].x = ov[d].x * corr + pe * vv.x;
            ov[d].y = ov[d].y * corr + pe * vv.y;
            ov[d].z = ov[d].z * corr + pe * vv.z;
            ov[d].w = ov[d].w * corr + pe * vv.w;
        }
        mM = nm;
    };

    if (am[b * Sz] > 0) accum(0);               // s0 term (global key 0)
    for (int j = tid; j <= tid + 2 * Wz; ++j) { // rel in [-W, W]
        int kp = c * Wz + j - Wz;
        if (kp < 1 || kp >= Sz) continue;
        if (am[b * Sz + kp] <= 0) continue;
        accum(kp);
    }

    float inv = 1.f / sS;
    float4* o4 = (float4*)(AO + ((size_t)(b * Sz + qrow)) * Dz + h * DHz);
    #pragma unroll
    for (int d = 0; d < 16; ++d)
        o4[d] = make_float4(ov[d].x * inv, ov[d].y * inv, ov[d].z * inv, ov[d].w * inv);
}

// =====================================================================
// Global attention for token 0: one (b,h) per block; 256 threads stride
// over S keys; LDS tree reduction + ds_add_f32 atomics for the output.
// =====================================================================
__global__ __launch_bounds__(256)
void global_attn_kernel(const float* __restrict__ QG, const float* __restrict__ KG,
                        const float* __restrict__ VG, const int* __restrict__ am,
                        float* __restrict__ AO)
{
    const int h = blockIdx.x % Hz;
    const int b = blockIdx.x / Hz;
    const int tid = threadIdx.x;
    const float scale = 0.125f;

    __shared__ float red[256];
    __shared__ float outAcc[DHz];

    const float4* q4 = (const float4*)(QG + ((size_t)(b * Sz)) * Dz + h * DHz);
    float4 qv[16];
    #pragma unroll
    for (int d = 0; d < 16; ++d) qv[d] = q4[d];

    float mM = -3.4e38f, sS = 0.f;
    float4 ov[16];
    #pragma unroll
    for (int d = 0; d < 16; ++d) ov[d] = make_float4(0.f, 0.f, 0.f, 0.f);

    for (int s = tid; s < Sz; s += 256) {
        if (am[b * Sz + s] <= 0) continue;
        const float4* k4 = (const float4*)(KG + ((size_t)(b * Sz + s)) * Dz + h * DHz);
        const float4* v4 = (const float4*)(VG + ((size_t)(b * Sz + s)) * Dz + h * DHz);
        float dot = 0.f;
        #pragma unroll
        for (int d = 0; d < 16; ++d) {
            float4 kk = k4[d];
            dot += qv[d].x * kk.x + qv[d].y * kk.y + qv[d].z * kk.z + qv[d].w * kk.w;
        }
        float sc = dot * scale;
        float nm = fmaxf(mM, sc);
        float corr = __expf(mM - nm);
        float pe   = __expf(sc - nm);
        sS = sS * corr + pe;
        #pragma unroll
        for (int d = 0; d < 16; ++d) {
            float4 vv = v4[d];
            ov[d].x = ov[d].x * corr + pe * vv.x;
            ov[d].y = ov[d].y * corr + pe * vv.y;
            ov[d].z = ov[d].z * corr + pe * vv.z;
            ov[d].w = ov[d].w * corr + pe * vv.w;
        }
        mM = nm;
    }

    red[tid] = mM; __syncthreads();
    for (int off = 128; off > 0; off >>= 1) {
        if (tid < off) red[tid] = fmaxf(red[tid], red[tid + off]);
        __syncthreads();
    }
    const float gmax = red[0]; __syncthreads();

    const float corr = __expf(mM - gmax);
    red[tid] = sS * corr; __syncthreads();
    for (int off = 128; off > 0; off >>= 1) {
        if (tid < off) red[tid] += red[tid + off];
        __syncthreads();
    }
    const float gsum = red[0];

    if (tid < DHz) outAcc[tid] = 0.f;
    __syncthreads();
    #pragma unroll
    for (int d = 0; d < 16; ++d) {
        atomicAdd(&outAcc[4 * d + 0], ov[d].x * corr);
        atomicAdd(&outAcc[4 * d + 1], ov[d].y * corr);
        atomicAdd(&outAcc[4 * d + 2], ov[d].z * corr);
        atomicAdd(&outAcc[4 * d + 3], ov[d].w * corr);
    }
    __syncthreads();
    if (tid < DHz)
        AO[((size_t)(b * Sz)) * Dz + h * DHz + tid] = outAcc[tid] / gsum;
}

// =====================================================================
// X = LayerNorm(X (+ R)) * g + b — one row (768) per block, 256 threads
// =====================================================================
__global__ __launch_bounds__(256)
void add_ln_kernel(float* __restrict__ X, const float* __restrict__ R,
                   const float* __restrict__ g, const float* __restrict__ bt)
{
    __shared__ float red[256];
    const int tid = threadIdx.x;
    const size_t base = (size_t)blockIdx.x * Dz;

    float v[3];
    float lsum = 0.f;
    #pragma unroll
    for (int i = 0; i < 3; ++i) {
        int c = tid + i * 256;
        float t = X[base + c];
        if (R) t += R[base + c];
        v[i] = t; lsum += t;
    }
    red[tid] = lsum; __syncthreads();
    for (int off = 128; off > 0; off >>= 1) {
        if (tid < off) red[tid] += red[tid + off];
        __syncthreads();
    }
    const float mean = red[0] * (1.f / (float)Dz);
    __syncthreads();

    float lsq = 0.f;
    #pragma unroll
    for (int i = 0; i < 3; ++i) { float d = v[i] - mean; lsq += d * d; }
    red[tid] = lsq; __syncthreads();
    for (int off = 128; off > 0; off >>= 1) {
        if (tid < off) red[tid] += red[tid + off];
        __syncthreads();
    }
    const float inv = rsqrtf(red[0] * (1.f / (float)Dz) + 1e-5f);

    #pragma unroll
    for (int i = 0; i < 3; ++i) {
        int c = tid + i * 256;
        X[base + c] = (v[i] - mean) * inv * g[c] + bt[c];
    }
}

// =====================================================================
// Embedding gather: X[b,s,:] = tok_emb[id] + pos_emb[s]
// =====================================================================
__global__ __launch_bounds__(256)
void embed_kernel(const int* __restrict__ ids, const float* __restrict__ tok,
                  const float* __restrict__ pos, float* __restrict__ X)
{
    const int row = blockIdx.x;
    const int s = row % Sz;
    const int id = ids[row];
    for (int c = threadIdx.x; c < Dz; c += 256)
        X[(size_t)row * Dz + c] = tok[(size_t)id * Dz + c] + pos[(size_t)s * Dz + c];
}

// =====================================================================
// Classifier: out[b,l] = X[b,0,:] . Wc[:,l] + bc[l]
// =====================================================================
__global__ __launch_bounds__(128)
void classifier_kernel(const float* __restrict__ X, const float* __restrict__ Wc,
                       const float* __restrict__ bc, float* __restrict__ out)
{
    const int t = threadIdx.x + blockIdx.x * blockDim.x;
    if (t >= Bz * NUM_LABELS) return;
    const int b = t / NUM_LABELS, l = t % NUM_LABELS;
    const float* xr = X + (size_t)b * Sz * Dz;   // row s=0
    float acc = bc[l];
    for (int d = 0; d < Dz; ++d) acc += xr[d] * Wc[d * NUM_LABELS + l];
    out[t] = acc;
}

// =====================================================================
// Host orchestration
// =====================================================================
extern "C" void kernel_launch(void* const* d_in, const int* in_sizes, int n_in,
                              void* d_out, int out_size, void* d_ws, size_t ws_size,
                              hipStream_t stream)
{
    (void)in_sizes; (void)n_in; (void)out_size; (void)ws_size;

    const int*   ids  = (const int*)d_in[0];
    const int*   am   = (const int*)d_in[1];
    const float* tok  = (const float*)d_in[2];
    const float* pos  = (const float*)d_in[3];
    const float* lneg = (const float*)d_in[4];
    const float* lneb = (const float*)d_in[5];
    const float* Wq   = (const float*)d_in[6];
    const float* bq   = (const float*)d_in[7];
    const float* Wk   = (const float*)d_in[8];
    const float* bk   = (const float*)d_in[9];
    const float* Wv   = (const float*)d_in[10];
    const float* bv   = (const float*)d_in[11];
    const float* Wqg  = (const float*)d_in[12];
    const float* bqg  = (const float*)d_in[13];
    const float* Wkg  = (const float*)d_in[14];
    const float* bkg  = (const float*)d_in[15];
    const float* Wvg  = (const float*)d_in[16];
    const float* bvg  = (const float*)d_in[17];
    const float* Wo   = (const float*)d_in[18];
    const float* bo   = (const float*)d_in[19];
    const float* ln1g = (const float*)d_in[20];
    const float* ln1b = (const float*)d_in[21];
    const float* W1   = (const float*)d_in[22];
    const float* b1   = (const float*)d_in[23];
    const float* W2   = (const float*)d_in[24];
    const float* b2   = (const float*)d_in[25];
    const float* ln2g = (const float*)d_in[26];
    const float* ln2b = (const float*)d_in[27];
    const float* Wc   = (const float*)d_in[28];
    const float* bc   = (const float*)d_in[29];

    float* ws = (float*)d_ws;
    float* X  = ws;            // B*S*D
    float* Q  = X  + BSD;
    float* Kq = Q  + BSD;
    float* V  = Kq + BSD;
    float* QG = V  + BSD;
    float* KG = QG + BSD;
    float* VG = KG + BSD;
    float* AO = VG + BSD;      // attention output / gemm2 output
    float* T1 = AO + BSD;      // B*S*FF scratch (attn-proj & FFN hidden)

    const int M = Bz * Sz;     // 4096
    const dim3 blk256(256);
    const dim3 gD (Dz  / TN, M / TM);   // N=768
    const dim3 gFF(FFz / TN, M / TM);   // N=3072

    embed_kernel<<<Bz * Sz, blk256, 0, stream>>>(ids, tok, pos, X);
    add_ln_kernel<<<Bz * Sz, blk256, 0, stream>>>(X, nullptr, lneg, lneb);

    for (int i = 0; i < Lz; ++i) {
        const size_t wo  = (size_t)i * Dz * Dz;
        const size_t bo_ = (size_t)i * Dz;
        const size_t w1o = (size_t)i * Dz * FFz;
        const size_t w2o = (size_t)i * FFz * Dz;

        gemm_wmma_kernel<<<gD, blk256, 0, stream>>>(X, Wq  + wo, bq  + bo_, Q,  M, Dz, Dz, 0);
        gemm_wmma_kernel<<<gD, blk256, 0, stream>>>(X, Wk  + wo, bk  + bo_, Kq, M, Dz, Dz, 0);
        gemm_wmma_kernel<<<gD, blk256, 0, stream>>>(X, Wv  + wo, bv  + bo_, V,  M, Dz, Dz, 0);
        gemm_wmma_kernel<<<gD, blk256, 0, stream>>>(X, Wqg + wo, bqg + bo_, QG, M, Dz, Dz, 0);
        gemm_wmma_kernel<<<gD, blk256, 0, stream>>>(X, Wkg + wo, bkg + bo_, KG, M, Dz, Dz, 0);
        gemm_wmma_kernel<<<gD, blk256, 0, stream>>>(X, Wvg + wo, bvg + bo_, VG, M, Dz, Dz, 0);

        sliding_attn_kernel<<<Bz * Hz * NCz, blk256, 0, stream>>>(Q, Kq, V, am, AO);
        global_attn_kernel <<<Bz * Hz,       blk256, 0, stream>>>(QG, KG, VG, am, AO);

        gemm_wmma_kernel<<<gD, blk256, 0, stream>>>(AO, Wo + wo, bo + bo_, T1, M, Dz, Dz, 0);
        add_ln_kernel<<<Bz * Sz, blk256, 0, stream>>>(X, T1, ln1g + bo_, ln1b + bo_);

        gemm_wmma_kernel<<<gFF, blk256, 0, stream>>>(X,  W1 + w1o, b1 + (size_t)i * FFz, T1, M, FFz, Dz, 1);
        gemm_wmma_kernel<<<gD,  blk256, 0, stream>>>(T1, W2 + w2o, b2 + bo_,             AO, M, Dz, FFz, 0);
        add_ln_kernel<<<Bz * Sz, blk256, 0, stream>>>(X, AO, ln2g + bo_, ln2b + bo_);
    }

    classifier_kernel<<<1, 128, 0, stream>>>(X, Wc, bc, (float*)d_out);
}
